// Block_SA_88210038325862
// MI455X (gfx1250) — matrix operand order repaired and checked by reading, not compiled, not run on hardware
//
#include <hip/hip_runtime.h>
#include <hip/hip_bf16.h>

// ---------------------------------------------------------------------------
// Fused attention block for MI455X (gfx1250, wave32, WMMA, async-to-LDS).
//   prep_kernel : conv1(1x1)+BN+ReLU -> Vt(bf16,[B,64,N]); sqrt(scale)*x -> Qb
//   attn_kernel : flash attention; 4 waves/block share K/V slabs staged in LDS
//                 by global_load_async_to_lds_b128 (double buffered, ASYNCcnt)
//   post_kernel : depthwise 3x3 + BN + ReLU + pointwise + BN + residual
// Shapes fixed by reference: B=8, C=64, H=W=64, N=4096.
// ---------------------------------------------------------------------------

typedef __bf16 bf16;
typedef __attribute__((ext_vector_type(16))) __bf16 v16bf;
typedef __attribute__((ext_vector_type(8)))  __bf16 v8bf;
typedef __attribute__((ext_vector_type(8)))  float  v8f;
typedef __attribute__((ext_vector_type(4)))  int    v4i;

#define B_   8
#define C64  64
#define N_   4096
#define EPS_ 1e-5f
// sqrt( log2(e) / sqrt(64) ) : folded once into Q (Q==K==x, so the product
// picks up the full softmax scale in the exp2 domain).
#define SQRT_SCALE 0.42466089f

#if __has_builtin(__builtin_amdgcn_exp2f)
#define EXP2F(x) __builtin_amdgcn_exp2f(x)
#else
#define EXP2F(x) exp2f(x)
#endif

static __device__ __forceinline__ v16bf cat8(v8bf lo, v8bf hi) {
  return __builtin_shufflevector(lo, hi, 0,1,2,3,4,5,6,7,8,9,10,11,12,13,14,15);
}

// Async global -> LDS copy of 16 bytes per lane (CDNA5, tracked by ASYNCcnt).
// ROCm clang types this builtin with int4 pointers: AS1 source, AS3 dest.
typedef __attribute__((address_space(1))) v4i* as1_v4p;
typedef __attribute__((address_space(3))) v4i* as3_v4p;

static __device__ __forceinline__ void async_ld16(const bf16* g, bf16* l) {
#if __has_builtin(__builtin_amdgcn_global_load_async_to_lds_b128)
  __builtin_amdgcn_global_load_async_to_lds_b128(
      (as1_v4p)(uintptr_t)g, (as3_v4p)(void*)l, 0, 0);
#else
  unsigned loff = (unsigned)(uintptr_t)(__attribute__((address_space(3))) void*)(void*)l;
  asm volatile("global_load_async_to_lds_b128 %0, %1, off"
               :: "v"(loff), "v"(g) : "memory");
#endif
}

static __device__ __forceinline__ void wait_async0() {
#if __has_builtin(__builtin_amdgcn_s_wait_asynccnt)
  __builtin_amdgcn_s_wait_asynccnt(0);
#else
  asm volatile("s_wait_asynccnt 0x0" ::: "memory");
#endif
}

// ---------------------------------------------------------------------------
// Kernel 1: conv1 (1x1) + BN1 + ReLU -> Vt bf16 [B][64][N]  (channel-major)
//           sqrt(scale) * x         -> Qb bf16 [B][N][64]  (position-major)
// ---------------------------------------------------------------------------
__global__ __launch_bounds__(256) void prep_kernel(
    const float* __restrict__ x,  const float* __restrict__ w1,
    const float* __restrict__ g1, const float* __restrict__ b1,
    const float* __restrict__ m1, const float* __restrict__ v1,
    bf16* __restrict__ Qb, bf16* __restrict__ Vt) {
  __shared__ float xs[64][64];   // [c][p]
  __shared__ float wsm[64][64];  // [o][c]
  __shared__ float sc[64], bi[64];
  const int tid = threadIdx.x;
  const int b  = blockIdx.x >> 6;
  const int nb = (blockIdx.x & 63) << 6;

  for (int idx = tid; idx < 4096; idx += 256) {
    int c = idx >> 6, p = idx & 63;
    xs[c][p]  = x[(b * 64 + c) * N_ + nb + p];
    wsm[c][p] = w1[idx];                       // wsm[o][cin]
  }
  if (tid < 64) {
    float inv = g1[tid] * rsqrtf(v1[tid] + EPS_);
    sc[tid] = inv;
    bi[tid] = b1[tid] - m1[tid] * inv;
  }
  __syncthreads();

  {
    const int p = tid & 63, og = tid >> 6;
    for (int k = 0; k < 16; ++k) {
      int o = og * 16 + k;
      float acc = 0.f;
      #pragma unroll
      for (int c = 0; c < 64; ++c) acc = fmaf(wsm[o][c], xs[c][p], acc);
      float val = fmaxf(fmaf(acc, sc[o], bi[o]), 0.f);
      Vt[(b * 64 + o) * N_ + nb + p] = (bf16)val;
    }
  }
  {
    const int c = tid & 63, pg = tid >> 6;
    for (int k = 0; k < 16; ++k) {
      int pp = pg * 16 + k;
      Qb[(b * N_ + nb + pp) * 64 + c] = (bf16)(xs[c][pp] * SQRT_SCALE);
    }
  }
}

// ---------------------------------------------------------------------------
// Kernel 2: flash attention. 4 waves per block, 16 queries per wave.
// Per 32-key step the block fetches one K slab (32x64 bf16) and one V slab
// (64x32 bf16) into double-buffered LDS with async loads; every wave runs
// 8x v_wmma_f32_16x16x32_bf16 against it. S^T = K*Q^T keeps the probability
// tile in A-matrix layout; O-rescale is skipped unless a row max increased.
// ---------------------------------------------------------------------------
__global__ __launch_bounds__(128) void attn_kernel(
    const bf16* __restrict__ Qb, const bf16* __restrict__ Vt,
    float* __restrict__ attn) {
  __shared__ alignas(64) bf16 ldsK[2][32][64];  // [stage][key][channel]
  __shared__ alignas(64) bf16 ldsV[2][64][32];  // [stage][channel][key]

  const int b    = blockIdx.x >> 6;            // 64 blocks (of 64 queries) per batch
  const int wv   = threadIdx.x >> 5;           // wave id: 4 waves per block
  const int ln   = threadIdx.x & 31;           // lane in wave
  const int row  = ((blockIdx.x & 63) * 4 + wv) * 16;
  const int half = ln >> 4;
  const int lq   = ln & 15;

  const bf16* kbase = Qb + (size_t)b * (N_ * 64);
  const bf16* vbase = Vt + (size_t)b * (64 * N_);

  // Q as WMMA B-matrix (32x16, K=d): lane = query column, loaded once.
  const bf16* qrow = Qb + (size_t)(b * N_ + row + lq) * 64;
  const v16bf qB0 = cat8(*(const v8bf*)(qrow + 0  + half * 8), *(const v8bf*)(qrow + 16 + half * 8));
  const v16bf qB1 = cat8(*(const v8bf*)(qrow + 32 + half * 8), *(const v8bf*)(qrow + 48 + half * 8));

  v8f o0 = {}, o1 = {}, o2 = {}, o3 = {};
  float m_run = -3.0e38f, l_run = 0.f;

  // 256 16B-chunks per slab; wave wv, instr i covers chunks wv*64 + i*32 + ln.
  auto issue_stage = [&](int col, int st) {
    #pragma unroll
    for (int i = 0; i < 2; ++i) {
      int chunk = wv * 64 + i * 32 + ln;
      int km = chunk >> 3, ks = chunk & 7;             // K: [m][d] rows of 128B
      async_ld16(kbase + (size_t)(col + km) * 64 + ks * 8, &ldsK[st][km][ks * 8]);
      int vd = chunk >> 2, vs = chunk & 3;             // V: [d][m] rows of 64B
      async_ld16(vbase + (size_t)vd * N_ + col + vs * 8, &ldsV[st][vd][vs * 8]);
    }
  };

  issue_stage(0, 0);

  for (int j = 0; j < N_ / 32; ++j) {
    const int cur = j & 1;
    wait_async0();        // my stage-j copies are done
    __syncthreads();      // everyone's are; everyone finished reading stage j-1
    if (j + 1 < N_ / 32) issue_stage((j + 1) * 32, cur ^ 1);  // overlap next slab

    // --- K tiles as A-matrix (16 keys x 32 channels, two d-steps each) ---
    v16bf a00 = cat8(*(const v8bf*)&ldsK[cur][lq][half * 8],
                     *(const v8bf*)&ldsK[cur][lq][16 + half * 8]);
    v16bf a01 = cat8(*(const v8bf*)&ldsK[cur][lq][32 + half * 8],
                     *(const v8bf*)&ldsK[cur][lq][48 + half * 8]);
    v16bf a10 = cat8(*(const v8bf*)&ldsK[cur][16 + lq][half * 8],
                     *(const v8bf*)&ldsK[cur][16 + lq][16 + half * 8]);
    v16bf a11 = cat8(*(const v8bf*)&ldsK[cur][16 + lq][32 + half * 8],
                     *(const v8bf*)&ldsK[cur][16 + lq][48 + half * 8]);

    // S^T tiles: M = key, N = query (C-layout lane = query column lq)
    v8f z = {};
    v8f st0 = __builtin_amdgcn_wmma_f32_16x16x32_bf16(false, a00, false, qB0, (short)0, z,   false, false);
    st0     = __builtin_amdgcn_wmma_f32_16x16x32_bf16(false, a01, false, qB1, (short)0, st0, false, false);
    v8f st1 = __builtin_amdgcn_wmma_f32_16x16x32_bf16(false, a10, false, qB0, (short)0, z,   false, false);
    st1     = __builtin_amdgcn_wmma_f32_16x16x32_bf16(false, a11, false, qB1, (short)0, st1, false, false);

    // --- V tiles as B-matrix (32 keys x 16 channels each, 4 d-tiles) ---
    v16bf vB0 = cat8(*(const v8bf*)&ldsV[cur][0 * 16 + lq][half * 8], *(const v8bf*)&ldsV[cur][0 * 16 + lq][16 + half * 8]);
    v16bf vB1 = cat8(*(const v8bf*)&ldsV[cur][1 * 16 + lq][half * 8], *(const v8bf*)&ldsV[cur][1 * 16 + lq][16 + half * 8]);
    v16bf vB2 = cat8(*(const v8bf*)&ldsV[cur][2 * 16 + lq][half * 8], *(const v8bf*)&ldsV[cur][2 * 16 + lq][16 + half * 8]);
    v16bf vB3 = cat8(*(const v8bf*)&ldsV[cur][3 * 16 + lq][half * 8], *(const v8bf*)&ldsV[cur][3 * 16 + lq][16 + half * 8]);

    // --- online softmax (exp2 domain; scale pre-folded into Q) ---
    float mx = -3.0e38f;
    #pragma unroll
    for (int v = 0; v < 8; ++v) mx = fmaxf(mx, fmaxf(st0[v], st1[v]));
    mx = fmaxf(mx, __shfl_xor(mx, 16, 32));

    const bool upd = __any(mx > m_run);
    const float m_new = fmaxf(m_run, mx);

    float rs = 0.f;
    v16bf pA;   // probability tile already in A-matrix layout (K = key index)
    #pragma unroll
    for (int v = 0; v < 8; ++v) {
      float p0 = EXP2F(st0[v] - m_new);
      float p1 = EXP2F(st1[v] - m_new);
      rs += p0 + p1;
      pA[v]     = (bf16)p0;
      pA[8 + v] = (bf16)p1;
    }
    rs += __shfl_xor(rs, 16, 32);

    if (upd) {            // wave-uniform branch: EXEC stays all-ones for WMMA
      const float alpha = EXP2F(m_run - m_new);
      m_run = m_new;
      l_run = fmaf(l_run, alpha, rs);
      const int abits = __float_as_int(alpha);
      #pragma unroll
      for (int v = 0; v < 8; ++v) {   // O-row n = v + 8*half; alpha lives on lane n
        float av = __int_as_float(__builtin_amdgcn_ds_bpermute((v + 8 * half) << 2, abits));
        o0[v] *= av; o1[v] *= av; o2[v] *= av; o3[v] *= av;
      }
    } else {
      l_run += rs;
    }

    o0 = __builtin_amdgcn_wmma_f32_16x16x32_bf16(false, pA, false, vB0, (short)0, o0, false, false);
    o1 = __builtin_amdgcn_wmma_f32_16x16x32_bf16(false, pA, false, vB1, (short)0, o1, false, false);
    o2 = __builtin_amdgcn_wmma_f32_16x16x32_bf16(false, pA, false, vB2, (short)0, o2, false, false);
    o3 = __builtin_amdgcn_wmma_f32_16x16x32_bf16(false, pA, false, vB3, (short)0, o3, false, false);
  }

  // final 1/l normalization and store: attn f32 [B][N][64]
  const int lbits = __float_as_int(l_run);
  float* abase = attn + (size_t)(b * N_ + row) * 64;
  #pragma unroll
  for (int v = 0; v < 8; ++v) {
    float lv  = __int_as_float(__builtin_amdgcn_ds_bpermute((v + 8 * half) << 2, lbits));
    float inv = 1.0f / lv;
    float* orow = abase + (size_t)(v + 8 * half) * 64 + lq;
    orow[0]  = o0[v] * inv;
    orow[16] = o1[v] * inv;
    orow[32] = o2[v] * inv;
    orow[48] = o3[v] * inv;
  }
}

// ---------------------------------------------------------------------------
// Kernel 3: depthwise 3x3 + BN2 + ReLU, pointwise 1x1 + BN3, + residual.
// ---------------------------------------------------------------------------
__global__ __launch_bounds__(256) void post_kernel(
    const float* __restrict__ attn, const float* __restrict__ x,
    const float* __restrict__ w2,
    const float* __restrict__ g2, const float* __restrict__ b2,
    const float* __restrict__ m2, const float* __restrict__ v2,
    const float* __restrict__ w3,
    const float* __restrict__ g3, const float* __restrict__ b3,
    const float* __restrict__ m3, const float* __restrict__ v3,
    float* __restrict__ out) {
  __shared__ float halo[100][64];
  __shared__ float t2[64][65];
  __shared__ float w3s[64][64];
  __shared__ float w2s[64][9];
  __shared__ float sc2[64], bi2[64], sc3[64], bi3[64];
  const int tid  = threadIdx.x;
  const int b    = blockIdx.x >> 6;
  const int tile = blockIdx.x & 63;
  const int th0  = (tile >> 3) << 3;
  const int tw0  = (tile & 7) << 3;

  for (int idx = tid; idx < 6400; idx += 256) {
    int c = idx & 63, pp = idx >> 6;
    int hh = th0 + pp / 10 - 1, ww = tw0 + pp % 10 - 1;
    float val = 0.f;
    if (hh >= 0 && hh < 64 && ww >= 0 && ww < 64)
      val = attn[(size_t)(b * N_ + hh * 64 + ww) * 64 + c];
    halo[pp][c] = val;
  }
  for (int idx = tid; idx < 4096; idx += 256) w3s[idx >> 6][idx & 63] = w3[idx];
  for (int idx = tid; idx < 576;  idx += 256) w2s[idx / 9][idx % 9]   = w2[idx];
  if (tid < 64) {
    float i2 = g2[tid] * rsqrtf(v2[tid] + EPS_);
    sc2[tid] = i2; bi2[tid] = b2[tid] - m2[tid] * i2;
    float i3 = g3[tid] * rsqrtf(v3[tid] + EPS_);
    sc3[tid] = i3; bi3[tid] = b3[tid] - m3[tid] * i3;
  }
  __syncthreads();

  {
    const int c = tid & 63, pg = tid >> 6;
    for (int k = 0; k < 16; ++k) {
      int p = pg * 16 + k;
      int ph = p >> 3, pw = p & 7;
      float acc = 0.f;
      #pragma unroll
      for (int dh = 0; dh < 3; ++dh)
        #pragma unroll
        for (int dw = 0; dw < 3; ++dw)
          acc = fmaf(halo[(ph + dh) * 10 + pw + dw][c], w2s[c][dh * 3 + dw], acc);
      t2[p][c] = fmaxf(fmaf(acc, sc2[c], bi2[c]), 0.f);
    }
  }
  __syncthreads();

  {
    const int p = tid & 63, og = tid >> 6;
    const int pos = (th0 + (p >> 3)) * 64 + (tw0 + (p & 7));
    for (int k = 0; k < 16; ++k) {
      int o = og * 16 + k;
      float acc = 0.f;
      #pragma unroll
      for (int c = 0; c < 64; ++c) acc = fmaf(w3s[o][c], t2[p][c], acc);
      float val = fmaf(acc, sc3[o], bi3[o]);
      size_t gi = (size_t)(b * 64 + o) * N_ + pos;
      out[gi] = val + x[gi];
    }
  }
}

// ---------------------------------------------------------------------------
extern "C" void kernel_launch(void* const* d_in, const int* in_sizes, int n_in,
                              void* d_out, int out_size, void* d_ws, size_t ws_size,
                              hipStream_t stream) {
  (void)in_sizes; (void)n_in; (void)out_size; (void)ws_size;
  const float* x  = (const float*)d_in[0];
  const float* w1 = (const float*)d_in[1];
  const float* g1 = (const float*)d_in[2];
  const float* b1 = (const float*)d_in[3];
  const float* m1 = (const float*)d_in[4];
  const float* v1 = (const float*)d_in[5];
  const float* w2 = (const float*)d_in[6];
  const float* g2 = (const float*)d_in[7];
  const float* b2 = (const float*)d_in[8];
  const float* m2 = (const float*)d_in[9];
  const float* v2 = (const float*)d_in[10];
  const float* w3 = (const float*)d_in[11];
  const float* g3 = (const float*)d_in[12];
  const float* b3 = (const float*)d_in[13];
  const float* m3 = (const float*)d_in[14];
  const float* v3 = (const float*)d_in[15];

  // workspace: Qb bf16 (4MB) | Vt bf16 (4MB) | attn f32 (8MB) = 16MB
  const size_t elems = (size_t)B_ * N_ * C64;
  bf16*  Qb   = (bf16*)d_ws;
  bf16*  Vt   = (bf16*)((char*)d_ws + elems * sizeof(bf16));
  float* attn = (float*)((char*)d_ws + elems * 2 * sizeof(bf16));

  prep_kernel<<<B_ * (N_ / 64), 256, 0, stream>>>(x, w1, g1, b1, m1, v1, Qb, Vt);
  attn_kernel<<<B_ * (N_ / 64), 128, 0, stream>>>(Qb, Vt, attn);
  post_kernel<<<B_ * 64, 256, 0, stream>>>(attn, x, w2, g2, b2, m2, v2,
                                           w3, g3, b3, m3, v3, (float*)d_out);
}
